// ProbAttention_84911503442551
// MI455X (gfx1250) — matrix operand order repaired and checked by reading, not compiled, not run on hardware
//
#include <hip/hip_runtime.h>

// ProbSparse attention (Informer) for MI455X / gfx1250.
// wave32, WMMA bf16, TDM (tensor_load_to_lds) double-buffered V staging,
// 4 waves/block flash-split over the key dimension with LDS merge.
// B=2, L=4096, H=8, D=64, FACTOR=5 -> U_part = u = 45.

#define BB 2
#define LL 4096
#define HH 8
#define DD 64
#define UU 45
#define UPAD 48      // 3 query tiles of 16
#define QTILES 3
#define NWAVES 4
#define NEGBIG (-1.0e30f)

typedef __attribute__((ext_vector_type(16))) __bf16 v16bf;
typedef __attribute__((ext_vector_type(8)))  float  v8f;
typedef __attribute__((ext_vector_type(4)))  unsigned int u32x4;
typedef __attribute__((ext_vector_type(8)))  int    i32x8;
typedef __attribute__((ext_vector_type(4)))  int    i32x4;

#if defined(__AMDGCN__) && __has_builtin(__builtin_amdgcn_tensor_load_to_lds)
#define HAVE_TDM 1
#else
#define HAVE_TDM 0
#endif

#if defined(__AMDGCN__) && __has_builtin(__builtin_amdgcn_s_wait_tensorcnt)
#define WAIT_TCNT(n) __builtin_amdgcn_s_wait_tensorcnt(n)
#else
#define WAIT_TCNT(n) do {} while (0)
#endif

union Frag16 {               // one WMMA 16-bit A or B operand (8 VGPRs/lane)
    v16bf v;
    unsigned short u16[16];
    unsigned int   u32[8];
};

__device__ __forceinline__ unsigned short f2bf(float x) {
    unsigned int u = __float_as_uint(x);
    return (unsigned short)((u + 0x7FFFu + ((u >> 16) & 1u)) >> 16);  // RNE
}
__device__ __forceinline__ unsigned int pack2bf(float lo, float hi) {
    return (unsigned int)f2bf(lo) | ((unsigned int)f2bf(hi) << 16);
}
__device__ __forceinline__ unsigned int mixhash(unsigned int x) {
    x ^= x >> 16; x *= 0x7feb352du;
    x ^= x >> 15; x *= 0x846ca68bu;
    x ^= x >> 16; return x;
}

// ---------------------------------------------------------------------------
// Stage one 32x64 fp32 V tile (row stride H*D=512 floats) into LDS with a
// 1-dword pad every 64 dwords (row stride 65 floats). TDM path builds a real
// 2D tile D# descriptor; fallback is a cooperative per-lane row copy.
// ---------------------------------------------------------------------------
__device__ __forceinline__ void stage_v_tile(float* dst, const float* __restrict__ V,
                                             int b, int h, int key0, int lane) {
    const float* src = V + ((size_t)(b * LL + key0) * HH + h) * DD;
#if HAVE_TDM
    (void)lane;
    unsigned long long va = (unsigned long long)(const void*)src;
    unsigned int lds_off  = (unsigned int)(unsigned long long)(const void*)dst;
    u32x4 g0;
    g0[0] = 1u;                                             // count=1, user desc
    g0[1] = lds_off;                                        // LDS byte address
    g0[2] = (unsigned int)(va & 0xFFFFFFFFu);               // global_addr[31:0]
    g0[3] = (unsigned int)((va >> 32) & 0x01FFFFFFu)        // global_addr[56:32]
          | (2u << 30);                                     // type = 2 (image)
    i32x8 g1 = {};
    g1[0] = (int)((2u << 16)                                // data_size = 4B
          |       (1u << 20)                                // pad_enable
          |       (5u << 22));                              // pad every 64 dwords
                                                            // pad_amount=0 -> 1 dword
    g1[1] = (int)(((unsigned)DD & 0xFFFFu) << 16);          // tensor_dim0 = 64
    g1[2] = (int)((32u & 0xFFFFu) << 16);                   // tensor_dim1 = 32
    g1[3] = (int)((unsigned)DD << 16);                      // tile_dim0 = 64
    g1[4] = 32;                                             // tile_dim1 = 32
    g1[5] = HH * DD;                                        // dim0 stride = 512 elems
    g1[6] = 0;
    g1[7] = 0;
    i32x4 gz = {};
#if defined(__clang_major__) && __clang_major__ >= 23
    i32x8 gz8 = {};
    __builtin_amdgcn_tensor_load_to_lds(g0, g1, gz, gz, gz8, 0);
#else
    __builtin_amdgcn_tensor_load_to_lds(g0, g1, gz, gz, 0);
#endif
#else
    const float* row = src + (size_t)lane * (HH * DD);
    float* d = dst + lane * 65;
#pragma unroll
    for (int i = 0; i < DD; i += 4) {
        float4 v = *(const float4*)(row + i);
        d[i] = v.x; d[i + 1] = v.y; d[i + 2] = v.z; d[i + 3] = v.w;
    }
#endif
}

// ---------------------------------------------------------------------------
// Kernel 1: sparsity measure M[b,h,q] = max_s(Q.K_s) - sum_s(Q.K_s)/L
// ---------------------------------------------------------------------------
__global__ void __launch_bounds__(256)
probM_kernel(const float* __restrict__ Q, const float* __restrict__ K,
             float* __restrict__ M) {
    int t = blockIdx.x * blockDim.x + threadIdx.x;
    if (t >= BB * HH * LL) return;
    int q  = t & (LL - 1);
    int bh = t >> 12;
    int h  = bh & (HH - 1);
    int b  = bh >> 3;

    const float* qrow = Q + (((size_t)b * LL + q) * HH + h) * DD;
    float4 qreg[DD / 4];
#pragma unroll
    for (int i = 0; i < DD / 4; ++i) qreg[i] = ((const float4*)qrow)[i];

    float mx = NEGBIG, sm = 0.0f;
    for (int s = 0; s < UU; ++s) {
        unsigned int r = mixhash((unsigned int)q * 0x9E3779B9u + (unsigned int)s + 1u);
        int key = (int)(r & (LL - 1));
        const float4* krow = (const float4*)(K + (((size_t)b * LL + key) * HH + h) * DD);
        float dot = 0.0f;
#pragma unroll
        for (int i = 0; i < DD / 4; ++i) {
            float4 kv = krow[i];
            dot = fmaf(qreg[i].x, kv.x, dot);
            dot = fmaf(qreg[i].y, kv.y, dot);
            dot = fmaf(qreg[i].z, kv.z, dot);
            dot = fmaf(qreg[i].w, kv.w, dot);
        }
        mx = fmaxf(mx, dot);
        sm += dot;
    }
    M[(size_t)bh * LL + q] = mx - sm * (1.0f / (float)LL);
}

// ---------------------------------------------------------------------------
// Kernel 2: per-(b,h) top-UPAD selection over L scores (masked block argmax).
// ---------------------------------------------------------------------------
__global__ void __launch_bounds__(256)
topk_kernel(const float* __restrict__ M, int* __restrict__ Mtop) {
    int bh = blockIdx.x;
    const float* m = M + (size_t)bh * LL;

    __shared__ float sv[256];
    __shared__ int   si[256];
    __shared__ unsigned int taken[LL / 32];

    for (int i = threadIdx.x; i < LL / 32; i += 256) taken[i] = 0u;
    __syncthreads();

    for (int r = 0; r < UPAD; ++r) {
        float best = NEGBIG;
        int   bi   = 0;
        for (int i = threadIdx.x; i < LL; i += 256) {
            if ((taken[i >> 5] >> (i & 31)) & 1u) continue;
            float v = m[i];
            if (v > best) { best = v; bi = i; }
        }
        sv[threadIdx.x] = best;
        si[threadIdx.x] = bi;
        __syncthreads();
#pragma unroll
        for (int off = 128; off > 0; off >>= 1) {
            if (threadIdx.x < off) {
                if (sv[threadIdx.x + off] > sv[threadIdx.x]) {
                    sv[threadIdx.x] = sv[threadIdx.x + off];
                    si[threadIdx.x] = si[threadIdx.x + off];
                }
            }
            __syncthreads();
        }
        if (threadIdx.x == 0) {
            int w = si[0];
            Mtop[bh * UPAD + r] = w;
            taken[w >> 5] |= (1u << (w & 31));
        }
        __syncthreads();
    }
}

// ---------------------------------------------------------------------------
// Kernel 3: flash attention over the 45 selected queries.
// 4 waves per block; wave w handles key tiles w, w+4, ... (32 tiles each),
// with TDM double-buffered V staging; partial (m,l,O) merged through LDS.
// ---------------------------------------------------------------------------
__global__ void __launch_bounds__(32 * NWAVES)
attn_wmma_kernel(const float* __restrict__ Q, const float* __restrict__ K,
                 const float* __restrict__ V, const int* __restrict__ mask,
                 const int* __restrict__ Mtop, float* __restrict__ out) {
    const int tid   = threadIdx.x;
    const int wave  = tid >> 5;
    const int lane  = tid & 31;
    const int qt    = blockIdx.x % QTILES;
    const int bh    = blockIdx.x / QTILES;
    const int h     = bh & (HH - 1);
    const int b     = bh >> 3;
    const bool lo16 = lane < 16;
    const int  col  = lane & 15;
    const float scale  = 0.125f;                 // 1/sqrt(64)
    const int  kdhalf  = lo16 ? 0 : 16;

    // per-wave P staging: [16 rows][32 keys] bf16, row stride 34 u16 (17 u32)
    __shared__ unsigned int ldsP[NWAVES][16 * 17];
    // per-wave double-buffered fp32 V tiles: [32 keys][64 d], row stride 65
    __shared__ float ldsVst[2][NWAVES][32 * 65];
    // cross-wave merge buffers
    __shared__ float cm[NWAVES][16], cl[NWAVES][16];
    __shared__ float cO[NWAVES][16][DD];

    // ---- load the Q A-fragments once (gathered through Mtop) ----
    int qr_for_A = qt * 16 + col;
    int qidx = Mtop[bh * UPAD + (qr_for_A < UPAD ? qr_for_A : UPAD - 1)];
    const float* qrow = Q + (((size_t)b * LL + qidx) * HH + h) * DD;

    Frag16 a0, a1;                               // k = 0..31 and 32..63
#pragma unroll
    for (int f = 0; f < 2; ++f) {
        int d0 = f * 32 + (lo16 ? 0 : 8);
        float4 r0 = *(const float4*)(qrow + d0);
        float4 r1 = *(const float4*)(qrow + d0 + 4);
        float4 r2 = *(const float4*)(qrow + d0 + 16);
        float4 r3 = *(const float4*)(qrow + d0 + 20);
        Frag16& a = f ? a1 : a0;
        a.u32[0] = pack2bf(r0.x, r0.y); a.u32[1] = pack2bf(r0.z, r0.w);
        a.u32[2] = pack2bf(r1.x, r1.y); a.u32[3] = pack2bf(r1.z, r1.w);
        a.u32[4] = pack2bf(r2.x, r2.y); a.u32[5] = pack2bf(r2.z, r2.w);
        a.u32[6] = pack2bf(r3.x, r3.y); a.u32[7] = pack2bf(r3.z, r3.w);
    }

    v8f o0 = {}, o1 = {}, o2 = {}, o3 = {};
    float mrow[8], lrow[8];
#pragma unroll
    for (int j = 0; j < 8; ++j) { mrow[j] = NEGBIG; lrow[j] = 0.0f; }

    unsigned short* ldsP16 = (unsigned short*)&ldsP[wave][0];
    float* stg[2] = { &ldsVst[0][wave][0], &ldsVst[1][wave][0] };

    const int NT = LL / 32;                      // 128 key tiles
    stage_v_tile(stg[0], V, b, h, wave * 32, lane);   // prologue DMA
    int cur = 0;

    for (int kt = wave; kt < NT; kt += NWAVES) {
        const int key0 = kt * 32;
        const int ktn  = kt + NWAVES;
        if (ktn < NT)
            stage_v_tile(stg[cur ^ 1], V, b, h, ktn * 32, lane);  // async next

        // prefetch next K tile rows into cache (global_prefetch_b8)
        if (ktn < NT)
            __builtin_prefetch(K + (((size_t)b * LL + ktn * 32 + lane) * HH + h) * DD, 0, 0);

        // ---- S = Q . K^T ; two 16-key subtiles, k-dim 64 in two chunks ----
        v8f sa = {}, sb = {};
#pragma unroll
        for (int s = 0; s < 2; ++s) {
            int key = key0 + s * 16 + col;
            const float* kr = K + (((size_t)b * LL + key) * HH + h) * DD;
            Frag16 bk0, bk1;
#pragma unroll
            for (int f = 0; f < 2; ++f) {
                const float* p = kr + f * 32 + (lo16 ? 0 : 8);   // d 8h..8h+7, 16+8h..16+8h+7
                float4 c0 = *(const float4*)(p);
                float4 c1 = *(const float4*)(p + 4);
                float4 c2 = *(const float4*)(p + 16);
                float4 c3 = *(const float4*)(p + 20);
                Frag16& bk = f ? bk1 : bk0;
                bk.u32[0] = pack2bf(c0.x, c0.y); bk.u32[1] = pack2bf(c0.z, c0.w);
                bk.u32[2] = pack2bf(c1.x, c1.y); bk.u32[3] = pack2bf(c1.z, c1.w);
                bk.u32[4] = pack2bf(c2.x, c2.y); bk.u32[5] = pack2bf(c2.z, c2.w);
                bk.u32[6] = pack2bf(c3.x, c3.y); bk.u32[7] = pack2bf(c3.z, c3.w);
            }
            v8f acc = {};
            acc = __builtin_amdgcn_wmma_f32_16x16x32_bf16(false, a0.v, false, bk0.v,
                                                          (short)0, acc, false, false);
            acc = __builtin_amdgcn_wmma_f32_16x16x32_bf16(false, a1.v, false, bk1.v,
                                                          (short)0, acc, false, false);
            if (s == 0) sa = acc; else sb = acc;
        }

        // ---- online softmax + write P into LDS (A-fragment friendly) ----
        int ma = mask[b * LL + key0 + col];
        int mb = mask[b * LL + key0 + 16 + col];
#pragma unroll
        for (int j = 0; j < 8; ++j) {
            int row = lo16 ? j : (j + 8);
            float va = ma ? sa[j] * scale : NEGBIG;
            float vb = mb ? sb[j] * scale : NEGBIG;
            float tm = fmaxf(va, vb);
#pragma unroll
            for (int d = 1; d < 16; d <<= 1)
                tm = fmaxf(tm, __shfl_xor(tm, d, 32));
            float mnew  = fmaxf(mrow[j], tm);
            float alpha = __expf(mrow[j] - mnew);
            float pa = __expf(va - mnew);
            float pb = __expf(vb - mnew);
            float rs = pa + pb;
#pragma unroll
            for (int d = 1; d < 16; d <<= 1)
                rs += __shfl_xor(rs, d, 32);
            lrow[j] = lrow[j] * alpha + rs;
            mrow[j] = mnew;
            o0[j] *= alpha; o1[j] *= alpha; o2[j] *= alpha; o3[j] *= alpha;
            ldsP16[row * 34 + col]      = f2bf(pa);
            ldsP16[row * 34 + col + 16] = f2bf(pb);
        }

        // ---- reload P as an A-fragment (16x32, k = key-in-tile) ----
        Frag16 pf;
        {
            int rbase = col * 17;
            int koff  = lo16 ? 0 : 4;
#pragma unroll
            for (int tI = 0; tI < 4; ++tI) {
                pf.u32[tI]     = ldsP[wave][rbase + koff + tI];
                pf.u32[4 + tI] = ldsP[wave][rbase + 8 + koff + tI];
            }
        }

        // ---- wait for this tile's TDM DMA, then O += P . V ----
#if HAVE_TDM
        if (ktn < NT) WAIT_TCNT(1); else WAIT_TCNT(0);
#endif
        const float* vt = stg[cur];
#pragma unroll
        for (int tI = 0; tI < 4; ++tI) {
            Frag16 vf;
            int d = tI * 16 + col;
#pragma unroll
            for (int i = 0; i < 16; ++i)
                vf.u16[i] = f2bf(vt[((lo16 ? 0 : 8) + (i < 8 ? i : i + 8)) * 65 + d]);
            v8f& oacc = (tI == 0) ? o0 : (tI == 1) ? o1 : (tI == 2) ? o2 : o3;
            oacc = __builtin_amdgcn_wmma_f32_16x16x32_bf16(false, pf.v, false, vf.v,
                                                           (short)0, oacc, false, false);
        }
        cur ^= 1;
    }

    // ---- publish per-wave partial (m, l, O) ----
#pragma unroll
    for (int j = 0; j < 8; ++j) {
        int row = lo16 ? j : (j + 8);
        if (col == 0) { cm[wave][row] = mrow[j]; cl[wave][row] = lrow[j]; }
        cO[wave][row][col]      = o0[j];
        cO[wave][row][col + 16] = o1[j];
        cO[wave][row][col + 32] = o2[j];
        cO[wave][row][col + 48] = o3[j];
    }
    __syncthreads();

    // ---- merge the 4 wave partials and scatter to out[b, qrank, h, d] ----
    for (int idx = tid; idx < 16 * DD; idx += 32 * NWAVES) {
        int row = idx >> 6;
        int d   = idx & (DD - 1);
        int qrank = qt * 16 + row;
        if (qrank >= UU) continue;
        float gm = cm[0][row];
#pragma unroll
        for (int w = 1; w < NWAVES; ++w) gm = fmaxf(gm, cm[w][row]);
        float lsum = 0.0f, val = 0.0f;
#pragma unroll
        for (int w = 0; w < NWAVES; ++w) {
            float e = __expf(cm[w][row] - gm);
            lsum += cl[w][row] * e;
            val  += cO[w][row][d] * e;
        }
        out[(((size_t)b * UU + qrank) * HH + h) * DD + d] =
            lsum > 0.0f ? val / lsum : 0.0f;
    }
}

// ---------------------------------------------------------------------------
extern "C" void kernel_launch(void* const* d_in, const int* in_sizes, int n_in,
                              void* d_out, int out_size, void* d_ws, size_t ws_size,
                              hipStream_t stream) {
    const float* Qp   = (const float*)d_in[0];
    const float* Kp   = (const float*)d_in[1];
    const float* Vp   = (const float*)d_in[2];
    const int*   mask = (const int*)d_in[3];
    float* out = (float*)d_out;

    float* Ms   = (float*)d_ws;                                   // B*H*L floats
    int*   Mtop = (int*)((char*)d_ws + sizeof(float) * BB * HH * LL);

    probM_kernel<<<(BB * HH * LL + 255) / 256, 256, 0, stream>>>(Qp, Kp, Ms);
    topk_kernel<<<BB * HH, 256, 0, stream>>>(Ms, Mtop);
    attn_wmma_kernel<<<BB * HH * QTILES, 32 * NWAVES, 0, stream>>>(Qp, Kp, Vp, mask,
                                                                   Mtop, out);
}
